// ElementwiseTensorProduct_50122268344455
// MI455X (gfx1250) — compile-verified, matrix operand order checked by reading, not run on hardware
//
#include <hip/hip_runtime.h>

typedef __attribute__((ext_vector_type(16))) _Float16 v16h;
typedef __attribute__((ext_vector_type(8)))  float    v8f;

#define Zdim 131072
#define Idim 32
#define Jdim 32
#define Kdim 64

// K-index mapping inside a 32-wide contraction chunk for 16-bit WMMA A/B frags
// (ISA 7.12.2: lanes 0-15 hold K=0..7 (v0-3) and 16..23 (v4-7); lanes 16-31 hold +8)
__device__ __forceinline__ int kmap(int h, int group) {
    return h + 8 * group + ((h >= 8) ? 8 : 0);
}

extern "C" __global__ __launch_bounds__(256)
void etp_wmma_kernel(const float* __restrict__ f1,
                     const float* __restrict__ f2,
                     const float* __restrict__ mix,
                     float* __restrict__ out) {
    // [32 chunks][4 ntiles][32 lanes][16 halfs] = 128 KB of pre-swizzled B frags
    extern __shared__ _Float16 Bs[];

    const int tid   = threadIdx.x;
    const int lane  = tid & 31;
    const int wave  = tid >> 5;
    const int group = lane >> 4;
    const int l16   = lane & 15;

    // ---- Stage B = mixing[k][i][j] (fp32) into LDS as fp16 WMMA-B fragments.
    // Fragment (c, n): contraction rows j=0..31 (i = c fixed), cols k = n*16..n*16+15.
    #pragma unroll
    for (int it = 0; it < 16; ++it) {
        int e = tid + it * 256;          // 0..4095 flat (c, n, l)
        int c = e >> 7;
        int n = (e >> 5) & 3;
        int l = e & 31;
        int g = l >> 4;
        int kcol = n * 16 + (l & 15);
        const float* mrow = mix + kcol * (Idim * Jdim) + c * Jdim;
        v16h tmp;
        #pragma unroll
        for (int h = 0; h < 16; ++h)
            tmp[h] = (_Float16)mrow[kmap(h, g)];
        *(v16h*)&Bs[(size_t)e * 16] = tmp;   // 32B aligned -> 2x ds_store_b128
    }
    __syncthreads();

    // ---- Each wave owns 32 rows of z (two 16-row WMMA subtiles).
    const long zbase = ((long)blockIdx.x * 8 + wave) * 32;

    _Float16 f1h[2][32];   // fp16 f1[z, i] for this lane's row, both subtiles
    v16h     f2h[2];       // fp16 f2[z, j] already in WMMA-A lane layout
    #pragma unroll
    for (int t = 0; t < 2; ++t) {
        const long z = zbase + t * 16 + l16;
        const float* f1r = f1 + z * Idim;
        const float* f2r = f2 + z * Jdim;
        #pragma unroll
        for (int c = 0; c < 32; ++c) f1h[t][c] = (_Float16)f1r[c];
        #pragma unroll
        for (int h = 0; h < 16; ++h) f2h[t][h] = (_Float16)f2r[kmap(h, group)];
    }

    v8f acc[2][4];
    #pragma unroll
    for (int t = 0; t < 2; ++t)
        #pragma unroll
        for (int n = 0; n < 4; ++n)
            acc[t][n] = (v8f){0.f,0.f,0.f,0.f,0.f,0.f,0.f,0.f};

    const v16h* Bf = (const v16h*)Bs;

    #pragma unroll
    for (int c = 0; c < 32; ++c) {
        // Four B fragments for this chunk from LDS (each 32B per lane).
        v16h b[4];
        #pragma unroll
        for (int n = 0; n < 4; ++n)
            b[n] = Bf[(c * 4 + n) * 32 + lane];

        #pragma unroll
        for (int t = 0; t < 2; ++t) {
            // A fragment = outer-product slice: a[h] = f1[z,c] * f2[z, j(h)]
            v16h a;
            _Float16 s = f1h[t][c];
            #pragma unroll
            for (int h = 0; h < 16; ++h) a[h] = s * f2h[t][h];
            #pragma unroll
            for (int n = 0; n < 4; ++n)
                acc[t][n] = __builtin_amdgcn_wmma_f32_16x16x32_f16(
                    /*neg_a=*/false, a, /*neg_b=*/false, b[n],
                    /*c_mod=*/(short)0, acc[t][n],
                    /*reuse_a=*/false, /*reuse_b=*/false);
        }
    }

    // ---- C/D layout: lane%16 = column, VGPR r = row r (+8 for lanes 16-31).
    #pragma unroll
    for (int t = 0; t < 2; ++t) {
        const long zrow0 = zbase + t * 16 + group * 8;
        #pragma unroll
        for (int n = 0; n < 4; ++n) {
            const int kc = n * 16 + l16;
            #pragma unroll
            for (int r = 0; r < 8; ++r)
                out[(zrow0 + r) * Kdim + kc] = acc[t][n][r];
        }
    }
}

extern "C" void kernel_launch(void* const* d_in, const int* in_sizes, int n_in,
                              void* d_out, int out_size, void* d_ws, size_t ws_size,
                              hipStream_t stream) {
    const float* f1  = (const float*)d_in[0];
    const float* f2  = (const float*)d_in[1];
    const float* mix = (const float*)d_in[2];
    float* out = (float*)d_out;
    (void)in_sizes; (void)n_in; (void)out_size; (void)d_ws; (void)ws_size;

    dim3 grid(Zdim / 256);   // 512 WGs; 8 waves/WG; 32 z-rows per wave
    dim3 block(256);
    size_t ldsBytes = 32u * 4u * 32u * 16u * sizeof(_Float16); // 128 KB
    etp_wmma_kernel<<<grid, block, ldsBytes, stream>>>(f1, f2, mix, out);
}
